// SelectiveSSM_57509612094226
// MI455X (gfx1250) — matrix operand order repaired and checked by reading, not compiled, not run on hardware
//
#include <hip/hip_runtime.h>
#include <hip/hip_bf16.h>

// ---------------------------------------------------------------------------
// Mamba selective-SSM forward for gfx1250 (MI455X), wave32.
//  - GEMMs: split-f16 (Markidis) on V_WMMA_F32_16X16X32_F16, double-buffered.
//  - Scan : 3-pass chunk-parallel, reproducing the reference's clamped
//           log-cumsum chunk math exactly (chunk = 32).
//  - Scan staging uses GLOBAL_LOAD_ASYNC_TO_LDS_B32 when available.
// ---------------------------------------------------------------------------

typedef __attribute__((ext_vector_type(16))) _Float16 v16h;
typedef __attribute__((ext_vector_type(8)))  _Float16 h8;
typedef __attribute__((ext_vector_type(8)))  float    v8f;

union V16 { v16h v; h8 p[2]; };

#define BSZ      2
#define LSEQ     2048
#define DMODEL   1024
#define DINNER   2048
#define DSTATE   16
#define MROWS    (BSZ * LSEQ)          // 4096 flattened (b,l) rows
#define CT       32                    // scan chunk (== reference CHUNK)
#define NCHUNK   (LSEQ / CT)           // 64

#define LOG1EM8  (-18.420680743952367f)   // ln(1e-8)

// ---- async global->LDS staging (CDNA5), with safe fallback -----------------
#if defined(__has_builtin)
#if __has_builtin(__builtin_amdgcn_global_load_async_to_lds_b32)
#define HAVE_ASYNC_B32 1
#endif
#if __has_builtin(__builtin_amdgcn_s_wait_asynccnt)
#define HAVE_WAIT_ASYNC 1
#endif
#endif

typedef __attribute__((address_space(1))) int as1_int;
typedef __attribute__((address_space(3))) int as3_int;

__device__ __forceinline__ void stage_b32(const float* g, float* l) {
#if defined(HAVE_ASYNC_B32)
    __builtin_amdgcn_global_load_async_to_lds_b32(
        (as1_int*)const_cast<float*>(g), (as3_int*)l, 0, 0);
#else
    *l = *g;
#endif
}

__device__ __forceinline__ void wait_stage() {
#if defined(HAVE_ASYNC_B32)
#if defined(HAVE_WAIT_ASYNC)
    __builtin_amdgcn_s_wait_asynccnt(0);
#else
    asm volatile("s_wait_asynccnt 0x0" ::: "memory");
#endif
#endif
}

__device__ __forceinline__ float silu_f(float v) {
    return v / (1.0f + __expf(-v));
}
__device__ __forceinline__ float softplus_f(float x) {
    return (x > 20.f) ? x : __logf(1.f + __expf(x));
}

// ---- GEMM tiling -----------------------------------------------------------
#define TM   128
#define TN   64
#define TK   32
#define KPAD 40                        // halves; 80B row stride -> conflict-free

// C[M,N] = A[M,K] @ B[K,N], split-f16 WMMA, double-buffered global loads.
// mode 0: out0[row*N + col] = v
// mode 1: col <  split -> out0[row*split + col]       = v        (x_in)
//         col >= split -> out1[row*split + col-split] = silu(v)  (silu(z))
__global__ __launch_bounds__(256)
void gemm_f16split(const float* __restrict__ A, const float* __restrict__ Bw,
                   float* __restrict__ out0, float* __restrict__ out1,
                   int M, int N, int K, int mode, int split)
{
    __shared__ __align__(16) _Float16 sAh[TM][KPAD];
    __shared__ __align__(16) _Float16 sAl[TM][KPAD];
    __shared__ __align__(16) _Float16 sBh[TN][KPAD];   // transposed: [n][k]
    __shared__ __align__(16) _Float16 sBl[TN][KPAD];

    const int tid  = threadIdx.x;
    const int lane = tid & 31;
    const int wv   = tid >> 5;                 // 8 waves
    const int rowBase = blockIdx.y * TM;
    const int colBase = blockIdx.x * TN;

    v8f acc[4] = {};

    // ISA 7.12.2 wave32 fragment addressing
    const int mrow = lane & 15;
    const int koA  = (lane >> 4) * 8;          // A: lanes16-31 hold K=8..15,24..31
    const int koB  = (lane >> 4) * 8;          // B: lanes16-31 hold K=8..15,24..31

    float4 ra[4];                              // prefetch registers
    float4 rb[2];
    auto gload = [&](int k0) {
        #pragma unroll
        for (int i = 0; i < 4; ++i) {
            const int linear = tid + i * 256;
            const int r  = linear >> 3;                       // 8 float4 / row
            const int kq = (linear & 7) * 4;
            ra[i] = *(const float4*)&A[(size_t)(rowBase + r) * K + k0 + kq];
        }
        #pragma unroll
        for (int i = 0; i < 2; ++i) {
            const int linear = tid + i * 256;
            const int kr = linear >> 4;                       // 16 float4 / k-row
            const int nq = (linear & 15) * 4;
            rb[i] = *(const float4*)&Bw[(size_t)(k0 + kr) * N + colBase + nq];
        }
    };

    gload(0);
    for (int k0 = 0; k0 < K; k0 += TK) {
        __syncthreads();               // previous tile's WMMA reads finished
        // ---- spill prefetched tile to LDS with fp32 -> hi/lo f16 split ----
        #pragma unroll
        for (int i = 0; i < 4; ++i) {
            const int linear = tid + i * 256;
            const int r  = linear >> 3;
            const int kq = (linear & 7) * 4;
            float t4[4] = { ra[i].x, ra[i].y, ra[i].z, ra[i].w };
            #pragma unroll
            for (int j = 0; j < 4; ++j) {
                _Float16 hi = (_Float16)t4[j];
                sAh[r][kq + j] = hi;
                sAl[r][kq + j] = (_Float16)(t4[j] - (float)hi);
            }
        }
        #pragma unroll
        for (int i = 0; i < 2; ++i) {
            const int linear = tid + i * 256;
            const int kr = linear >> 4;
            const int nq = (linear & 15) * 4;
            float t4[4] = { rb[i].x, rb[i].y, rb[i].z, rb[i].w };
            #pragma unroll
            for (int j = 0; j < 4; ++j) {
                _Float16 hi = (_Float16)t4[j];
                sBh[nq + j][kr] = hi;
                sBl[nq + j][kr] = (_Float16)(t4[j] - (float)hi);
            }
        }
        __syncthreads();
        if (k0 + TK < K) gload(k0 + TK);   // overlap next loads with WMMA

        V16 ah, al;
        ah.p[0] = *(const h8*)&sAh[wv * 16 + mrow][koA];
        ah.p[1] = *(const h8*)&sAh[wv * 16 + mrow][16 + koA];
        al.p[0] = *(const h8*)&sAl[wv * 16 + mrow][koA];
        al.p[1] = *(const h8*)&sAl[wv * 16 + mrow][16 + koA];

        #pragma unroll
        for (int nt = 0; nt < 4; ++nt) {
            const int nc = nt * 16 + mrow;
            V16 bh, bl;
            bh.p[0] = *(const h8*)&sBh[nc][koB];
            bh.p[1] = *(const h8*)&sBh[nc][koB + 16];
            bl.p[0] = *(const h8*)&sBl[nc][koB];
            bl.p[1] = *(const h8*)&sBl[nc][koB + 16];
            acc[nt] = __builtin_amdgcn_wmma_f32_16x16x32_f16(
                false, al.v, false, bh.v, (short)0, acc[nt], false, false);
            acc[nt] = __builtin_amdgcn_wmma_f32_16x16x32_f16(
                false, ah.v, false, bl.v, (short)0, acc[nt], false, false);
            acc[nt] = __builtin_amdgcn_wmma_f32_16x16x32_f16(
                false, ah.v, false, bh.v, (short)0, acc[nt], false, false);
        }
    }

    // ---- epilogue: VGPR r -> (M = r + 8*(lane>=16), N = lane&15) ----
    const int nOff = lane & 15;
    const int mOff = (lane >> 4) * 8;
    #pragma unroll
    for (int nt = 0; nt < 4; ++nt) {
        #pragma unroll
        for (int r = 0; r < 8; ++r) {
            const int grow = rowBase + wv * 16 + mOff + r;
            const int gcol = colBase + nt * 16 + nOff;
            const float v  = acc[nt][r];
            if (mode == 0) {
                out0[(size_t)grow * N + gcol] = v;
            } else {
                if (gcol < split) out0[(size_t)grow * split + gcol] = v;
                else              out1[(size_t)grow * split + (gcol - split)] = silu_f(v);
            }
        }
    }
}

// ---- depthwise causal conv (width 4) + bias + SiLU -------------------------
__global__ __launch_bounds__(256)
void conv_silu(const float* __restrict__ xin, const float* __restrict__ w,
               const float* __restrict__ bias, float* __restrict__ xc)
{
    const int idx = blockIdx.x * 256 + threadIdx.x;     // over B*L*DINNER
    const int d   = idx & (DINNER - 1);
    const int l   = (idx >> 11) & (LSEQ - 1);
    const int b   = idx >> 22;
    float acc = bias[d];
    #pragma unroll
    for (int j = 0; j < 4; ++j) {
        const int ll = l - 3 + j;
        if (ll >= 0)
            acc = fmaf(w[d * 4 + j], xin[((size_t)(b * LSEQ + ll)) * DINNER + d], acc);
    }
    xc[(size_t)idx] = silu_f(acc);
}

// ---- x_proj: per row, 33 dot-products of length DINNER ---------------------
__global__ __launch_bounds__(256)
void xproj(const float* __restrict__ xc, const float* __restrict__ W,
           float* __restrict__ Bm, float* __restrict__ Cm, float* __restrict__ dtr)
{
    __shared__ float srow[DINNER];
    const int row  = blockIdx.x;
    const int tid  = threadIdx.x;
    const int lane = tid & 31;
    const int wv   = tid >> 5;
    for (int i = tid; i < DINNER; i += 256) srow[i] = xc[(size_t)row * DINNER + i];
    __syncthreads();
    for (int o = wv; o < 2 * DSTATE + 1; o += 8) {
        float s = 0.f;
        for (int k = lane; k < DINNER; k += 32)
            s = fmaf(srow[k], W[(size_t)k * (2 * DSTATE + 1) + o], s);
        #pragma unroll
        for (int off = 16; off > 0; off >>= 1) s += __shfl_down(s, off, 32);
        if (lane == 0) {
            if      (o < DSTATE)     Bm[row * DSTATE + o] = s;
            else if (o < 2 * DSTATE) Cm[row * DSTATE + (o - DSTATE)] = s;
            else                     dtr[row] = s;
        }
    }
}

// ===========================================================================
// Chunk-parallel selective scan (reference-exact clamped log-cumsum math).
//   la       = max(delta*A_s, ln 1e-8)              (== log(clamp(dA)))
//   cum     += la                                    (cum <= 0)
//   inv      = exp(-max(cum, -20))
//   csum    += dBu * inv
//   state    = exp(cum) * (h0 + csum)
// Pass1: per (b,chunk,d): P_s = exp(cum_31), S_s = exp(cum_31)*csum_31
// Pass2: per (b,d): h <- P*h + S across chunks; stores h0 per chunk (in Sb)
// Pass3: per (b,chunk,d): replay with h0, y = (sum_s C*state + u*D)*silu(z)
// ===========================================================================

__global__ __launch_bounds__(256)
void scan_pass1(const float* __restrict__ xc,  const float* __restrict__ Bm,
                const float* __restrict__ dtr, const float* __restrict__ A_log,
                const float* __restrict__ dt_w, const float* __restrict__ dt_b,
                float* __restrict__ Pb, float* __restrict__ Sb)
{
    const int tid = threadIdx.x;
    const int d   = blockIdx.x * 256 + tid;
    const int c   = blockIdx.y;
    const int b   = blockIdx.z;

    __shared__ float sB[CT][DSTATE];
    __shared__ float sdt[CT];
    {
        const int rowBase = (b * LSEQ + c * CT) * DSTATE;
        for (int i = tid; i < CT * DSTATE; i += 256) stage_b32(&Bm[rowBase + i], &sB[0][0] + i);
        for (int i = tid; i < CT; i += 256)          stage_b32(&dtr[b * LSEQ + c * CT + i], &sdt[i]);
        wait_stage();
    }
    __syncthreads();

    float As[DSTATE], cum[DSTATE], csum[DSTATE];
    #pragma unroll
    for (int s = 0; s < DSTATE; ++s) {
        As[s]   = -__expf(A_log[d * DSTATE + s]);
        cum[s]  = 0.f;
        csum[s] = 0.f;
    }
    const float dtw = dt_w[d], dtb = dt_b[d];

    for (int tt = 0; tt < CT; ++tt) {
        const size_t off = ((size_t)(b * LSEQ + c * CT + tt)) * DINNER + d;
        const float u     = xc[off];
        const float delta = softplus_f(fmaf(sdt[tt], dtw, dtb));
        const float du    = delta * u;
        #pragma unroll
        for (int s = 0; s < DSTATE; ++s) {
            const float la = fmaxf(delta * As[s], LOG1EM8);
            cum[s] += la;
            const float inv = __expf(-fmaxf(cum[s], -20.f));
            csum[s] = fmaf(du * sB[tt][s], inv, csum[s]);
        }
    }
    const size_t base = ((size_t)(b * NCHUNK + c) * DSTATE) * DINNER + d;
    #pragma unroll
    for (int s = 0; s < DSTATE; ++s) {
        const float e1 = __expf(cum[s]);           // cum <= 0, no upper clamp needed
        Pb[base + (size_t)s * DINNER] = e1;
        Sb[base + (size_t)s * DINNER] = e1 * csum[s];
    }
}

__global__ __launch_bounds__(256)
void scan_pass2(const float* __restrict__ Pb, float* __restrict__ Sb)
{
    const int idx = blockIdx.x * 256 + threadIdx.x;   // 0 .. BSZ*DINNER-1
    const int b = idx >> 11;
    const int d = idx & (DINNER - 1);
    float h[DSTATE];
    #pragma unroll
    for (int s = 0; s < DSTATE; ++s) h[s] = 0.f;
    for (int c = 0; c < NCHUNK; ++c) {
        const size_t base = ((size_t)(b * NCHUNK + c) * DSTATE) * DINNER + d;
        #pragma unroll
        for (int s = 0; s < DSTATE; ++s) {
            const size_t a = base + (size_t)s * DINNER;
            const float p  = Pb[a];
            const float sv = Sb[a];
            Sb[a] = h[s];                           // chunk's initial state
            h[s]  = fmaf(p, h[s], sv);
        }
    }
}

__global__ __launch_bounds__(256)
void scan_pass3(const float* __restrict__ xc,  const float* __restrict__ sz,
                const float* __restrict__ Bm,  const float* __restrict__ Cm,
                const float* __restrict__ dtr, const float* __restrict__ A_log,
                const float* __restrict__ Dp,  const float* __restrict__ dt_w,
                const float* __restrict__ dt_b, const float* __restrict__ H0,
                float* __restrict__ yf)
{
    const int tid = threadIdx.x;
    const int d   = blockIdx.x * 256 + tid;
    const int c   = blockIdx.y;
    const int b   = blockIdx.z;

    __shared__ float sB[CT][DSTATE];
    __shared__ float sC[CT][DSTATE];
    __shared__ float sdt[CT];
    {
        const int rowBase = (b * LSEQ + c * CT) * DSTATE;
        for (int i = tid; i < CT * DSTATE; i += 256) {
            stage_b32(&Bm[rowBase + i], &sB[0][0] + i);
            stage_b32(&Cm[rowBase + i], &sC[0][0] + i);
        }
        for (int i = tid; i < CT; i += 256) stage_b32(&dtr[b * LSEQ + c * CT + i], &sdt[i]);
        wait_stage();
    }
    __syncthreads();

    float As[DSTATE], cum[DSTATE], csum[DSTATE], h0[DSTATE];
    const size_t base = ((size_t)(b * NCHUNK + c) * DSTATE) * DINNER + d;
    #pragma unroll
    for (int s = 0; s < DSTATE; ++s) {
        As[s]   = -__expf(A_log[d * DSTATE + s]);
        cum[s]  = 0.f;
        csum[s] = 0.f;
        h0[s]   = H0[base + (size_t)s * DINNER];
    }
    const float Dv = Dp[d], dtw = dt_w[d], dtb = dt_b[d];

    for (int tt = 0; tt < CT; ++tt) {
        const size_t off = ((size_t)(b * LSEQ + c * CT + tt)) * DINNER + d;
        const float u     = xc[off];
        const float delta = softplus_f(fmaf(sdt[tt], dtw, dtb));
        const float du    = delta * u;
        float acc = 0.f;
        #pragma unroll
        for (int s = 0; s < DSTATE; ++s) {
            const float la = fmaxf(delta * As[s], LOG1EM8);
            cum[s] += la;
            const float inv = __expf(-fmaxf(cum[s], -20.f));
            csum[s] = fmaf(du * sB[tt][s], inv, csum[s]);
            const float st = __expf(cum[s]) * (h0[s] + csum[s]);
            acc = fmaf(sC[tt][s], st, acc);
        }
        yf[off] = (acc + u * Dv) * sz[off];
    }
}

// ---------------------------------------------------------------------------
extern "C" void kernel_launch(void* const* d_in, const int* in_sizes, int n_in,
                              void* d_out, int out_size, void* d_ws, size_t ws_size,
                              hipStream_t stream) {
    (void)in_sizes; (void)n_in; (void)out_size; (void)ws_size;
    const float* x    = (const float*)d_in[0];   // (B,L,DMODEL)
    const float* w_in = (const float*)d_in[1];   // (DMODEL, 2*DINNER)
    const float* cw   = (const float*)d_in[2];   // (DINNER,1,4)
    const float* cb   = (const float*)d_in[3];   // (DINNER)
    const float* xpw  = (const float*)d_in[4];   // (DINNER, 33)
    const float* dtw  = (const float*)d_in[5];   // (1, DINNER)
    const float* dtb  = (const float*)d_in[6];   // (DINNER)
    const float* alog = (const float*)d_in[7];   // (DINNER, 16)
    const float* Dp   = (const float*)d_in[8];   // (DINNER)
    const float* wout = (const float*)d_in[9];   // (DINNER, DMODEL)
    float* out = (float*)d_out;                  // (B,L,DMODEL)

    // workspace layout (floats); Pbuf and yf alias xin (dead after conv/pass2)
    float* ws   = (float*)d_ws;
    float* xin  = ws;                                    // MROWS*DINNER
    float* sz   = xin  + (size_t)MROWS * DINNER;         // MROWS*DINNER
    float* xc   = sz   + (size_t)MROWS * DINNER;         // MROWS*DINNER
    float* Bmat = xc   + (size_t)MROWS * DINNER;         // MROWS*DSTATE
    float* Cmat = Bmat + (size_t)MROWS * DSTATE;         // MROWS*DSTATE
    float* dtr  = Cmat + (size_t)MROWS * DSTATE;         // MROWS
    float* Sbuf = dtr  + (size_t)MROWS;                  // BSZ*NCHUNK*DSTATE*DINNER
    float* Pbuf = xin;                                   // alias (x_in dead post-conv)
    float* yf   = xin;                                   // alias (P dead post-pass2)

    // 1) xz = x @ in_proj_w ; split -> x_in, silu(z)
    gemm_f16split<<<dim3((2 * DINNER) / TN, MROWS / TM), 256, 0, stream>>>(
        x, w_in, xin, sz, MROWS, 2 * DINNER, DMODEL, /*mode=*/1, /*split=*/DINNER);

    // 2) depthwise causal conv + SiLU
    conv_silu<<<(MROWS * DINNER) / 256, 256, 0, stream>>>(xin, cw, cb, xc);

    // 3) x_proj -> B, C, dt scalar
    xproj<<<MROWS, 256, 0, stream>>>(xc, xpw, Bmat, Cmat, dtr);

    // 4) chunk-parallel selective scan (3 passes), fused (+u*D)*silu(z)
    scan_pass1<<<dim3(DINNER / 256, NCHUNK, BSZ), 256, 0, stream>>>(
        xc, Bmat, dtr, alog, dtw, dtb, Pbuf, Sbuf);
    scan_pass2<<<(BSZ * DINNER) / 256, 256, 0, stream>>>(Pbuf, Sbuf);
    scan_pass3<<<dim3(DINNER / 256, NCHUNK, BSZ), 256, 0, stream>>>(
        xc, sz, Bmat, Cmat, dtr, alog, Dp, dtw, dtb, Sbuf, yf);

    // 5) out = yf @ out_proj_w
    gemm_f16split<<<dim3(DMODEL / TN, MROWS / TM), 256, 0, stream>>>(
        yf, wout, out, nullptr, MROWS, DMODEL, DINNER, /*mode=*/0, /*split=*/0);
}
